// EncoderLayer_45930380263715
// MI455X (gfx1250) — matrix operand order repaired and checked by reading, not compiled, not run on hardware
//
#include <hip/hip_runtime.h>
#include <cstddef>
#include <cstdint>

#define D_MODEL 512
#define N_HEADS 8
#define DK      64
#define D_FF    2048
#define BATCH   2
#define SEQ     2048
#define MROWS   (BATCH * SEQ)   // 4096

typedef __attribute__((ext_vector_type(16))) __bf16 v16bf;
typedef __attribute__((ext_vector_type(8)))  float  v8f;

union FragBF { v16bf v; unsigned int u[8]; };

__device__ __forceinline__ unsigned short f2bf(float f) {
    union { float f; unsigned int u; } x; x.f = f;
    unsigned int r = x.u + 0x7FFFu + ((x.u >> 16) & 1u);  // round-to-nearest-even
    return (unsigned short)(r >> 16);
}

__device__ __forceinline__ v8f wmma_bf16(const FragBF& a, const FragBF& b, v8f c) {
    return __builtin_amdgcn_wmma_f32_16x16x32_bf16(
        false, a.v, false, b.v, (short)0, c, false, false);
}

__device__ __forceinline__ float red_max16(float v) {
    for (int off = 1; off < 16; off <<= 1) v = fmaxf(v, __shfl_xor(v, off, 32));
    return v;
}
__device__ __forceinline__ float red_sum16(float v) {
    for (int off = 1; off < 16; off <<= 1) v += __shfl_xor(v, off, 32);
    return v;
}
__device__ __forceinline__ float red_sum32(float v) {
    for (int off = 1; off < 32; off <<= 1) v += __shfl_xor(v, off, 32);
    return v;
}

// async copy 16B global -> LDS (GV mode, per-lane LDS dest), tracked by ASYNCcnt
__device__ __forceinline__ void async_b128(unsigned lds_byte_off, const void* gptr) {
    asm volatile("global_load_async_to_lds_b128 %0, %1, off"
                 :: "v"(lds_byte_off), "v"((unsigned long long)gptr)
                 : "memory");
}
__device__ __forceinline__ void wait_async0() {
    asm volatile("s_wait_asynccnt 0x0" ::: "memory");
}

// ---------------------------------------------------------------- cast f32->bf16
__global__ void cast_f32_bf16(const float* __restrict__ in,
                              unsigned short* __restrict__ out, int n) {
    for (int i = blockIdx.x * blockDim.x + threadIdx.x; i < n;
         i += gridDim.x * blockDim.x)
        out[i] = f2bf(in[i]);
}

// ---------------------------------------------------------------- generic WMMA GEMM
// C[M,N] = A[M,K](bf16) * B[K,N](bf16) + bias (+resid) (relu?)
// Block tile 128x128, 8 waves in 4(M)x2(N) grid, each wave 32x64 (8 wmma / k-step).
// Double-buffered LDS; A tile via async DMA, B tile transposed in flight.
#define GBM 128
#define GBN 128
#define GBK 32
#define ASTR 40   // As row stride in bf16 (80B, 16B-aligned rows for async b128)
#define BSTR 34   // Bs row stride in bf16 (transposed [n][k])

template<int RELU, int RESID, int OUTF, int OUTB>
__global__ __launch_bounds__(256) void gemm_bf16(
    const unsigned short* __restrict__ A, const unsigned short* __restrict__ B,
    const float* __restrict__ bias, const float* __restrict__ resid,
    float* __restrict__ outF, unsigned short* __restrict__ outB,
    int M, int N, int K)
{
    __shared__ alignas(16) unsigned short As[2][GBM * ASTR];  // [m][k]
    __shared__ alignas(16) unsigned short Bs[2][GBN * BSTR];  // [n][k] (transposed)

    const int tid  = threadIdx.x;
    const int wave = tid >> 5, lane = tid & 31;
    const int half = lane >> 4, l16 = lane & 15;
    const int wm = wave >> 1, wn = wave & 1;
    const int bm = blockIdx.x * GBM;
    const int bn = blockIdx.y * GBN;

    auto stage = [&](int buf, int k0) {
        // A tile 128x32: async DMA, 2 x 16B chunks per thread
        unsigned aB = (unsigned)(uintptr_t)&As[buf][0];
        for (int it = 0; it < 2; ++it) {
            int cid = it * 256 + tid;      // 512 chunks
            int r = cid >> 2;              // 0..127
            int c = (cid & 3) * 8;         // 0,8,16,24
            async_b128(aB + (unsigned)(r * (ASTR * 2) + c * 2),
                       A + (size_t)(bm + r) * K + k0 + c);
        }
        // B tile 32x128, transposed into Bs[n][k]: 2 x 8 bf16 per thread
        for (int it = 0; it < 2; ++it) {
            int cid = it * 256 + tid;      // 512 chunks
            int kr = cid >> 4;             // 0..31
            int c  = (cid & 15) * 8;       // 0..120
            uint4 d = *(const uint4*)(B + (size_t)(k0 + kr) * N + bn + c);
            unsigned int w[4] = {d.x, d.y, d.z, d.w};
            for (int j = 0; j < 4; ++j) {
                Bs[buf][(c + 2 * j    ) * BSTR + kr] = (unsigned short)(w[j]);
                Bs[buf][(c + 2 * j + 1) * BSTR + kr] = (unsigned short)(w[j] >> 16);
            }
        }
    };

    v8f acc[2][4];
    for (int i = 0; i < 2; ++i)
        for (int j = 0; j < 4; ++j)
            acc[i][j] = (v8f){0.f,0.f,0.f,0.f,0.f,0.f,0.f,0.f};

    const int nk = K / GBK;
    stage(0, 0);
    for (int ki = 0; ki < nk; ++ki) {
        wait_async0();      // this wave's DMA for tile ki resident
        __syncthreads();    // all waves' stores/DMA for tile ki visible
        if (ki + 1 < nk) stage((ki + 1) & 1, (ki + 1) * GBK);  // overlap next DMA

        const unsigned int* AsU = (const unsigned int*)As[ki & 1];
        const unsigned int* BsU = (const unsigned int*)Bs[ki & 1];
        FragBF afr[2], bfr[4];
        for (int i = 0; i < 8; ++i) {
            int ka2 = (i & 3) + (i >> 2) * 8 + half * 4;  // A k-order / 2
            int kb2 = (i & 3) + (i >> 2) * 8 + half * 4;  // B k-order / 2
            afr[0].u[i] = AsU[(wm * 32      + l16) * (ASTR / 2) + ka2];
            afr[1].u[i] = AsU[(wm * 32 + 16 + l16) * (ASTR / 2) + ka2];
            bfr[0].u[i] = BsU[(wn * 64      + l16) * (BSTR / 2) + kb2];
            bfr[1].u[i] = BsU[(wn * 64 + 16 + l16) * (BSTR / 2) + kb2];
            bfr[2].u[i] = BsU[(wn * 64 + 32 + l16) * (BSTR / 2) + kb2];
            bfr[3].u[i] = BsU[(wn * 64 + 48 + l16) * (BSTR / 2) + kb2];
        }
        for (int mi = 0; mi < 2; ++mi)
            for (int ni = 0; ni < 4; ++ni)
                acc[mi][ni] = wmma_bf16(afr[mi], bfr[ni], acc[mi][ni]);
    }

    // epilogue: C layout m = r + half*8, n = l16 (branch-free, templated)
    for (int mi = 0; mi < 2; ++mi)
        for (int ni = 0; ni < 4; ++ni) {
            int gn = bn + wn * 64 + ni * 16 + l16;
            float bv = bias[gn];
            for (int r = 0; r < 8; ++r) {
                int gm = bm + wm * 32 + mi * 16 + r + half * 8;
                float v = acc[mi][ni][r] + bv;
                if (RELU)  v = fmaxf(v, 0.f);
                size_t idx = (size_t)gm * N + gn;
                if (RESID) v += resid[idx];
                if (OUTF)  outF[idx] = v;
                if (OUTB)  outB[idx] = f2bf(v);
            }
        }
}

// ---------------------------------------------------------------- flash attention
// grid.x = B*H (16), grid.y = SEQ/128. 8 waves/WG, wave owns 16 query rows.
// Double-buffered K/V staging; K tile via async DMA, V transposed in flight.
#define KSTR 72   // Ks row stride in bf16 (144B, 16B-aligned rows for async b128)

__global__ __launch_bounds__(256) void attn_kernel(
    const unsigned short* __restrict__ Q, const unsigned short* __restrict__ Kk,
    const unsigned short* __restrict__ V, const int* __restrict__ mask,
    unsigned short* __restrict__ ctx)
{
    __shared__ alignas(16) unsigned short Ks[2][32 * KSTR];   // [key][d]
    __shared__ alignas(16) unsigned short Vt[2][64 * 34];     // [d][key] stride 34
    __shared__ alignas(16) unsigned short Pl[8][16 * 32];     // per-wave P (row-major)

    const int tid  = threadIdx.x;
    const int wave = tid >> 5, lane = tid & 31;
    const int half = lane >> 4, l16 = lane & 15;
    const int b = blockIdx.x >> 3, h = blockIdx.x & 7;
    const int qr0 = blockIdx.y * 128 + wave * 16;

    auto stage = [&](int buf, int kb) {
        // K block (32 keys x 64 d) via async DMA: one 16B chunk per thread
        unsigned kB = (unsigned)(uintptr_t)&Ks[buf][0];
        int kr = tid >> 3;             // 0..31
        int c  = (tid & 7) * 8;        // 0..56
        async_b128(kB + (unsigned)(kr * (KSTR * 2) + c * 2),
                   Kk + ((size_t)(b * SEQ + kb + kr)) * D_MODEL + h * DK + c);
        // V block transposed: Vt[d][key]
        uint4 d = *(const uint4*)(V + ((size_t)(b * SEQ + kb + kr)) * D_MODEL + h * DK + c);
        unsigned int w[4] = {d.x, d.y, d.z, d.w};
        for (int j = 0; j < 4; ++j) {
            Vt[buf][(c + 2 * j    ) * 34 + kr] = (unsigned short)(w[j]);
            Vt[buf][(c + 2 * j + 1) * 34 + kr] = (unsigned short)(w[j] >> 16);
        }
    };

    // Q fragments (A layout): lane m = l16, d split into [0,32) and [32,64)
    FragBF qf[2];
    {
        size_t qbase = ((size_t)(b * SEQ + qr0 + l16)) * D_MODEL + h * DK;
        for (int i = 0; i < 8; ++i) {
            int ka = (i & 3) * 2 + (i >> 2) * 16 + half * 8;
            qf[0].u[i] = *(const unsigned int*)(Q + qbase + ka);
            qf[1].u[i] = *(const unsigned int*)(Q + qbase + 32 + ka);
        }
    }

    float mrow[8], lrow[8];
    v8f acc[4];
    for (int r = 0; r < 8; ++r) { mrow[r] = -1e30f; lrow[r] = 0.f; }
    for (int t = 0; t < 4; ++t) acc[t] = (v8f){0.f,0.f,0.f,0.f,0.f,0.f,0.f,0.f};

    const int nb = SEQ / 32;
    stage(0, 0);
    for (int kbi = 0; kbi < nb; ++kbi) {
        const int kb = kbi * 32;
        wait_async0();
        __syncthreads();
        if (kbi + 1 < nb) stage((kbi + 1) & 1, kb + 32);   // overlap next DMA

        const unsigned int* KsU = (const unsigned int*)Ks[kbi & 1];
        const unsigned int* VtU = (const unsigned int*)Vt[kbi & 1];

        // scores: two 16x16 tiles (key chunks), each = 2 chained WMMAs over d=64
        v8f s[2];
        for (int ch = 0; ch < 2; ++ch) {
            int krow = ch * 16 + l16;  // this lane's key (B-operand lane = column)
            FragBF kf0, kf1;
            for (int i = 0; i < 8; ++i) {
                int kb16 = (i & 3) + (i >> 2) * 8 + half * 4;   // table-order k / 2
                kf0.u[i] = KsU[krow * (KSTR / 2) + kb16];       // d in [0,32)
                kf1.u[i] = KsU[krow * (KSTR / 2) + 16 + kb16];  // d in [32,64)
            }
            v8f sc = (v8f){0.f,0.f,0.f,0.f,0.f,0.f,0.f,0.f};
            sc = wmma_bf16(qf[0], kf0, sc);
            sc = wmma_bf16(qf[1], kf1, sc);
            sc = sc * 0.125f;                              // 1/sqrt(64)
            int mk = mask[b * SEQ + kb + ch * 16 + l16];
            if (mk == 0)
                for (int r = 0; r < 8; ++r) sc[r] = -1e9f;
            s[ch] = sc;
        }

        // online softmax (per row = r + half*8; reduce over 16 lanes of same half)
        float bmax[8], rsum[8], fac[8];
        for (int r = 0; r < 8; ++r)
            bmax[r] = red_max16(fmaxf(s[0][r], s[1][r]));
        for (int r = 0; r < 8; ++r) {
            float mnew = fmaxf(mrow[r], bmax[r]);
            fac[r] = __expf(mrow[r] - mnew);
            float p0 = __expf(s[0][r] - mnew);
            float p1 = __expf(s[1][r] - mnew);
            s[0][r] = p0; s[1][r] = p1;
            rsum[r] = p0 + p1;
            mrow[r] = mnew;
        }
        for (int r = 0; r < 8; ++r) {
            lrow[r] = lrow[r] * fac[r] + red_sum16(rsum[r]);
            for (int t = 0; t < 4; ++t) acc[t][r] *= fac[r];
        }

        // P: C-layout -> per-wave LDS (row-major 16x32) -> A-layout fragment
        for (int ch = 0; ch < 2; ++ch)
            for (int r = 0; r < 8; ++r)
                Pl[wave][(r + half * 8) * 32 + ch * 16 + l16] = f2bf(s[ch][r]);
        asm volatile("s_wait_dscnt 0x0" ::: "memory");

        FragBF pf;
        {
            const unsigned int* PlU = (const unsigned int*)Pl[wave];
            for (int i = 0; i < 8; ++i) {
                int ka = (i & 3) * 2 + (i >> 2) * 16 + half * 8;
                pf.u[i] = PlU[l16 * 16 + (ka >> 1)];
            }
        }
        // ctx += P(16x32) x V(32x64) : 4 d-tiles
        for (int t = 0; t < 4; ++t) {
            FragBF vf;
            int dcol = t * 16 + l16;
            for (int i = 0; i < 8; ++i)
                vf.u[i] = VtU[dcol * 17 + (i & 3) + (i >> 2) * 8 + half * 4];
            acc[t] = wmma_bf16(pf, vf, acc[t]);
        }
    }

    // normalize and write ctx -> [B,S,D_MODEL] (heads re-merged)
    for (int r = 0; r < 8; ++r) {
        float inv = 1.f / (lrow[r] + 1e-20f);
        int qrow = qr0 + r + half * 8;
        size_t base = ((size_t)(b * SEQ + qrow)) * D_MODEL + h * DK;
        for (int t = 0; t < 4; ++t)
            ctx[base + t * 16 + l16] = f2bf(acc[t][r] * inv);
    }
}

// ---------------------------------------------------------------- layernorm (wave/row)
__global__ __launch_bounds__(256) void ln_kernel(
    const float* __restrict__ in, const float* __restrict__ g,
    const float* __restrict__ be, float* __restrict__ outF,
    unsigned short* __restrict__ outB, int rows)
{
    const int wave = threadIdx.x >> 5, lane = threadIdx.x & 31;
    const int row = blockIdx.x * 8 + wave;
    if (row >= rows) return;
    const float4* p = (const float4*)(in + (size_t)row * D_MODEL);
    float4 v[4];
    float s = 0.f, sq = 0.f;
    for (int j = 0; j < 4; ++j) {
        v[j] = p[j * 32 + lane];
        s  += v[j].x + v[j].y + v[j].z + v[j].w;
        sq += v[j].x * v[j].x + v[j].y * v[j].y + v[j].z * v[j].z + v[j].w * v[j].w;
    }
    s  = red_sum32(s);
    sq = red_sum32(sq);
    float mu   = s * (1.f / D_MODEL);
    float var  = sq * (1.f / D_MODEL) - mu * mu;
    float rstd = rsqrtf(var + 1e-5f);
    for (int j = 0; j < 4; ++j) {
        int d0 = (j * 32 + lane) * 4;
        float4 o;
        o.x = (v[j].x - mu) * rstd * g[d0 + 0] + be[d0 + 0];
        o.y = (v[j].y - mu) * rstd * g[d0 + 1] + be[d0 + 1];
        o.z = (v[j].z - mu) * rstd * g[d0 + 2] + be[d0 + 2];
        o.w = (v[j].w - mu) * rstd * g[d0 + 3] + be[d0 + 3];
        ((float4*)(outF + (size_t)row * D_MODEL))[j * 32 + lane] = o;
        if (outB) {
            size_t b0 = (size_t)row * D_MODEL + d0;
            outB[b0 + 0] = f2bf(o.x); outB[b0 + 1] = f2bf(o.y);
            outB[b0 + 2] = f2bf(o.z); outB[b0 + 3] = f2bf(o.w);
        }
    }
}

// ---------------------------------------------------------------- host orchestration
extern "C" void kernel_launch(void* const* d_in, const int* in_sizes, int n_in,
                              void* d_out, int out_size, void* d_ws, size_t ws_size,
                              hipStream_t stream) {
    const float* x    = (const float*)d_in[0];
    const int*   mask = (const int*)  d_in[1];
    const float* Wq = (const float*)d_in[2];  const float* bq = (const float*)d_in[3];
    const float* Wk = (const float*)d_in[4];  const float* bk = (const float*)d_in[5];
    const float* Wv = (const float*)d_in[6];  const float* bv = (const float*)d_in[7];
    const float* Wo = (const float*)d_in[8];  const float* bo = (const float*)d_in[9];
    const float* W1 = (const float*)d_in[10]; const float* b1 = (const float*)d_in[11];
    const float* W2 = (const float*)d_in[12]; const float* b2 = (const float*)d_in[13];
    const float* g1 = (const float*)d_in[14]; const float* be1 = (const float*)d_in[15];
    const float* g2 = (const float*)d_in[16]; const float* be2 = (const float*)d_in[17];

    char* ws = (char*)d_ws;
    size_t off = 0;
    auto alloc = [&](size_t bytes) -> void* {
        void* p = ws + off;
        off = (off + bytes + 255) & ~(size_t)255;
        return p;
    };
    unsigned short* Xb   = (unsigned short*)alloc((size_t)MROWS * D_MODEL * 2);
    unsigned short* Wqb  = (unsigned short*)alloc((size_t)D_MODEL * D_MODEL * 2);
    unsigned short* Wkb  = (unsigned short*)alloc((size_t)D_MODEL * D_MODEL * 2);
    unsigned short* Wvb  = (unsigned short*)alloc((size_t)D_MODEL * D_MODEL * 2);
    unsigned short* Wob  = (unsigned short*)alloc((size_t)D_MODEL * D_MODEL * 2);
    unsigned short* W1b  = (unsigned short*)alloc((size_t)D_MODEL * D_FF * 2);
    unsigned short* W2b  = (unsigned short*)alloc((size_t)D_FF * D_MODEL * 2);
    unsigned short* Qb   = (unsigned short*)alloc((size_t)MROWS * D_MODEL * 2);
    unsigned short* Kb   = (unsigned short*)alloc((size_t)MROWS * D_MODEL * 2);
    unsigned short* Vb   = (unsigned short*)alloc((size_t)MROWS * D_MODEL * 2);
    unsigned short* CTXb = (unsigned short*)alloc((size_t)MROWS * D_MODEL * 2);
    unsigned short* X1b  = (unsigned short*)alloc((size_t)MROWS * D_MODEL * 2);
    unsigned short* Hb   = (unsigned short*)alloc((size_t)MROWS * D_FF * 2);
    float* attnf = (float*)alloc((size_t)MROWS * D_MODEL * 4);
    float* x1f   = (float*)alloc((size_t)MROWS * D_MODEL * 4);
    float* ff2f  = (float*)alloc((size_t)MROWS * D_MODEL * 4);

    // casts
    cast_f32_bf16<<<1024, 256, 0, stream>>>(x,  Xb,  MROWS * D_MODEL);
    cast_f32_bf16<<<512,  256, 0, stream>>>(Wq, Wqb, D_MODEL * D_MODEL);
    cast_f32_bf16<<<512,  256, 0, stream>>>(Wk, Wkb, D_MODEL * D_MODEL);
    cast_f32_bf16<<<512,  256, 0, stream>>>(Wv, Wvb, D_MODEL * D_MODEL);
    cast_f32_bf16<<<512,  256, 0, stream>>>(Wo, Wob, D_MODEL * D_MODEL);
    cast_f32_bf16<<<1024, 256, 0, stream>>>(W1, W1b, D_MODEL * D_FF);
    cast_f32_bf16<<<1024, 256, 0, stream>>>(W2, W2b, D_FF * D_MODEL);

    // QKV projections (bf16 out only)
    dim3 gQKV(MROWS / GBM, D_MODEL / GBN);
    gemm_bf16<0,0,0,1><<<gQKV, 256, 0, stream>>>(Xb, Wqb, bq, nullptr, nullptr, Qb,
                                                 MROWS, D_MODEL, D_MODEL);
    gemm_bf16<0,0,0,1><<<gQKV, 256, 0, stream>>>(Xb, Wkb, bk, nullptr, nullptr, Kb,
                                                 MROWS, D_MODEL, D_MODEL);
    gemm_bf16<0,0,0,1><<<gQKV, 256, 0, stream>>>(Xb, Wvb, bv, nullptr, nullptr, Vb,
                                                 MROWS, D_MODEL, D_MODEL);

    // flash attention -> CTXb
    attn_kernel<<<dim3(BATCH * N_HEADS, SEQ / 128), 256, 0, stream>>>(
        Qb, Kb, Vb, mask, CTXb);

    // attn_out = ctx @ Wo + bo + x (residual), f32
    gemm_bf16<0,1,1,0><<<gQKV, 256, 0, stream>>>(CTXb, Wob, bo, x, attnf, nullptr,
                                                 MROWS, D_MODEL, D_MODEL);
    // x1 = LN(attnf)
    ln_kernel<<<MROWS / 8, 256, 0, stream>>>(attnf, g1, be1, x1f, X1b, MROWS);

    // ff hidden = relu(x1 @ W1 + b1), bf16
    dim3 gFF1(MROWS / GBM, D_FF / GBN);
    gemm_bf16<1,0,0,1><<<gFF1, 256, 0, stream>>>(X1b, W1b, b1, nullptr, nullptr, Hb,
                                                 MROWS, D_FF, D_MODEL);
    // ff out = h @ W2 + b2 + x1 (residual), f32
    gemm_bf16<0,1,1,0><<<gQKV, 256, 0, stream>>>(Hb, W2b, b2, x1f, ff2f, nullptr,
                                                 MROWS, D_MODEL, D_FF);
    // out = LN(ff2f)
    ln_kernel<<<MROWS / 8, 256, 0, stream>>>(ff2f, g2, be2, (float*)d_out,
                                             nullptr, MROWS);
}
